// MultiHeadTargetAttention_10136122819077
// MI455X (gfx1250) — matrix operand order repaired and checked by reading, not compiled, not run on hardware
//
#include <hip/hip_runtime.h>
#include <hip/hip_bf16.h>

typedef __attribute__((ext_vector_type(16))) _Float16 v16h;
typedef __attribute__((ext_vector_type(8)))  _Float16 v8h;
typedef __attribute__((ext_vector_type(8)))  float    v8f;

#if defined(__has_builtin) && __has_builtin(__builtin_amdgcn_sched_barrier)
#define SCHED_FENCE() __builtin_amdgcn_sched_barrier(0)
#else
#define SCHED_FENCE() ((void)0)
#endif

constexpr int L  = 200;   // history length
constexpr int D  = 128;   // model dim
constexpr int LP = 224;   // L padded to multiple of 32 (WMMA K)
constexpr int HT_STRIDE = 232; // transposed-history row stride (halves); 464B = 29*16 -> 16B-aligned rows

__global__ __launch_bounds__(256) void mhta_kernel(
    const float* __restrict__ target,   // [B,128]
    const float* __restrict__ hist,     // [B,200,128]
    const int*   __restrict__ mask,     // [B,200]
    const float* __restrict__ Wq,       // [128,128]
    const float* __restrict__ Wk,       // [128,128]
    const float* __restrict__ Wv,       // [128,128]
    const float* __restrict__ Wo,       // [128,128]
    float* __restrict__ outp)           // [B,128]
{
    __shared__ __align__(32) _Float16 sh_hist [LP][D];        // row-major f16 history (rows >= 200 garbage; their scores are never used)
    __shared__ __align__(32) _Float16 sh_histT[D][HT_STRIDE]; // transposed f16 history; cols [200,224) zeroed
    __shared__ __align__(32) _Float16 sh_qk  [16][D];         // qk rows 0..3 valid (scaled by 1/sqrt(32))
    __shared__ __align__(32) _Float16 sh_attn[16][LP];        // attn rows 0..3 valid; cols [200,224) zeroed
    __shared__ __align__(16) float sh_scores[4][LP];
    __shared__ __align__(16) float sh_q[D];
    __shared__ __align__(16) float sh_tgt[D];
    __shared__ __align__(16) float sh_wh[4][D];
    __shared__ __align__(16) float sh_out[D];

    const int b    = blockIdx.x;
    const int tid  = threadIdx.x;
    const int lane = tid & 31;
    // wave index is uniform across the wave: pin it to an SGPR so loops/branches on it
    // are scalar (s_cbranch) and EXEC is provably all-1s at every WMMA.
    const int wave = __builtin_amdgcn_readfirstlane(tid >> 5);

    // ---------- Phase A: stage target + history (f32 -> f16, row-major and transposed) ----------
    if (tid < D) sh_tgt[tid] = target[(size_t)b * D + tid];

    const float* hb = hist + (size_t)b * L * D;
    // 200*128 floats = 6400 float4; 25 per thread. Within a wave all lanes share row l.
    for (int i = 0; i < 25; ++i) {
        int idx = tid + i * 256;
        int l   = idx >> 5;
        int j0  = (idx & 31) * 4;
        float4 hv = *(const float4*)(hb + l * D + j0);
        _Float16 h0 = (_Float16)hv.x, h1 = (_Float16)hv.y, h2 = (_Float16)hv.z, h3 = (_Float16)hv.w;
        sh_hist[l][j0+0] = h0; sh_hist[l][j0+1] = h1; sh_hist[l][j0+2] = h2; sh_hist[l][j0+3] = h3;
        sh_histT[j0+0][l] = h0; sh_histT[j0+1][l] = h1; sh_histT[j0+2][l] = h2; sh_histT[j0+3][l] = h3;
    }
    // zero transposed-history K padding l in [200,224) so 0*pad stays 0 in the wh GEMM
    for (int idx = tid; idx < D * (LP - L); idx += 256) {
        int j = idx / (LP - L);
        int l = L + idx % (LP - L);
        sh_histT[j][l] = (_Float16)0.0f;
    }
    __syncthreads();

    // ---------- Phase B: q[a] = sum_j Wq[a,j] * tgt[j] ----------
    if (tid < D) {
        float acc = 0.f;
        const float4* wr = (const float4*)(Wq + tid * D);
        const float4* tg = (const float4*)sh_tgt;
        for (int j = 0; j < D / 4; ++j) {
            float4 w = wr[j], t = tg[j];
            acc += w.x * t.x + w.y * t.y + w.z * t.z + w.w * t.w;
        }
        sh_q[tid] = acc;
    }
    __syncthreads();

    // ---------- Phase C: qk[h][j] = (1/sqrt(32)) * sum_d q[h*32+d] * Wk[h*32+d, j] ----------
    if (tid < D) {
        const float inv_scale = 0.17677669529663687f; // 1/sqrt(HEAD_DIM)
        const int j = tid;
        for (int h = 0; h < 4; ++h) {
            float acc = 0.f;
            for (int d = 0; d < 32; ++d)                    // coalesced across lanes (j contiguous)
                acc += sh_q[h * 32 + d] * Wk[(h * 32 + d) * D + j];
            sh_qk[h][j] = (_Float16)(acc * inv_scale);
        }
    }
    __syncthreads();

    // ---------- Phase D (WMMA): scores[h][l] ; M=h(16), K=j(128), N=l(224) ----------
    // A = sh_qk (rows >=4 only pollute ignored D rows); B columns are rows of sh_hist.
    {
        const int m   = lane & 15;
        const int kbA = (lane < 16) ? 0 : 8;
        const int kbB = (lane < 16) ? 0 : 8;
        v16h afrag[4];
        for (int s = 0; s < 4; ++s) {
            v8h lo = *(const v8h*)&sh_qk[m][32 * s + kbA];
            v8h hi = *(const v8h*)&sh_qk[m][32 * s + kbA + 16];
            v16h a;
            for (int i = 0; i < 8; ++i) { a[i] = lo[i]; a[i + 8] = hi[i]; }
            afrag[s] = a;
        }
        for (int t = wave; t < 14; t += 8) {      // scalar loop bound: s_cbranch, EXEC untouched
            const int lcol = t * 16 + m;
            // issue ALL K-step B-fragment loads first ...
            v8h blo[4], bhi[4];
            for (int s = 0; s < 4; ++s) {
                blo[s] = *(const v8h*)&sh_hist[lcol][32 * s + kbB];
                bhi[s] = *(const v8h*)&sh_hist[lcol][32 * s + kbB + 16];
            }
            SCHED_FENCE();  // ... keep them hoisted above the WMMA chain (progressive dscnt waits)
            v8f c = {};
            for (int s = 0; s < 4; ++s) {
                v16h bf;
                for (int i = 0; i < 8; ++i) { bf[i] = blo[s][i]; bf[i + 8] = bhi[s][i]; }
                c = __builtin_amdgcn_wmma_f32_16x16x32_f16(false, afrag[s], false, bf,
                                                           (short)0, c, false, false);
            }
            if (lane < 16) {
                for (int r = 0; r < 4; ++r) sh_scores[r][t * 16 + lane] = c[r];
            }
        }
    }
    __syncthreads();

    // ---------- Phase E: masked softmax per head (wave w handles head w), branchless gather ----------
    if (wave < 4) {
        const int h = wave;
        const int* mb = mask + (size_t)b * L;
        float v[7];
        float mx = -3.0e38f;
        for (int i = 0; i < 7; ++i) {
            int l  = lane + 32 * i;
            int lc = (l < L) ? l : (L - 1);        // clamp: loads always unconditional & in-bounds
            int   mv   = mb[lc];
            float sraw = sh_scores[h][lc];
            float sv   = (l < L && mv != 0) ? sraw : -3.0e38f;  // v_cndmask, no exec branching
            v[i] = sv;
            mx = fmaxf(mx, sv);
        }
        for (int off = 16; off > 0; off >>= 1) mx = fmaxf(mx, __shfl_xor(mx, off, 32));
        float sum = 0.f;
        for (int i = 0; i < 7; ++i) { float e = __expf(v[i] - mx); v[i] = e; sum += e; }
        for (int off = 16; off > 0; off >>= 1) sum += __shfl_xor(sum, off, 32);
        float inv = 1.0f / sum;
        for (int i = 0; i < 7; ++i) {
            int l = lane + 32 * i;                 // covers all of [0,224): masked/pad lanes wrote exp->0
            sh_attn[h][l] = (_Float16)(v[i] * inv);
        }
    }
    __syncthreads();

    // ---------- Phase F (WMMA): wh[h][j] = sum_l attn[h,l]*hist[l,j] ; M=h, K=l(224), N=j(128) ----------
    // A = sh_attn ; B columns are rows of sh_histT (contiguous in l).
    {
        const int m   = lane & 15;
        const int kbA = (lane < 16) ? 0 : 8;
        const int kbB = (lane < 16) ? 0 : 8;
        const int jcol = wave * 16 + m;            // 8 waves cover 8 N-tiles exactly
        // issue ALL fragment loads (7 K-steps of A and B) first ...
        v16h afrag[7];
        for (int s = 0; s < 7; ++s) {
            v8h alo = *(const v8h*)&sh_attn[m][32 * s + kbA];
            v8h ahi = *(const v8h*)&sh_attn[m][32 * s + kbA + 16];
            v16h a;
            for (int i = 0; i < 8; ++i) { a[i] = alo[i]; a[i + 8] = ahi[i]; }
            afrag[s] = a;
        }
        v8h blo[7], bhi[7];
        for (int s = 0; s < 7; ++s) {
            blo[s] = *(const v8h*)&sh_histT[jcol][32 * s + kbB];
            bhi[s] = *(const v8h*)&sh_histT[jcol][32 * s + kbB + 16];
        }
        SCHED_FENCE();  // ... then run the 7-deep WMMA chain over them
        v8f c = {};
        for (int s = 0; s < 7; ++s) {
            v16h bf;
            for (int i = 0; i < 8; ++i) { bf[i] = blo[s][i]; bf[i + 8] = bhi[s][i]; }
            c = __builtin_amdgcn_wmma_f32_16x16x32_f16(false, afrag[s], false, bf,
                                                       (short)0, c, false, false);
        }
        if (lane < 16) {
            for (int r = 0; r < 4; ++r) sh_wh[r][jcol] = c[r];
        }
    }
    __syncthreads();

    // ---------- Phase G: out[a] = sum_j wh[a/32][j] * Wv[a,j] ----------
    if (tid < D) {
        const int a = tid, h = a >> 5;
        float acc = 0.f;
        const float4* wr = (const float4*)(Wv + a * D);
        const float4* wh = (const float4*)sh_wh[h];
        for (int j = 0; j < D / 4; ++j) {
            float4 w = wr[j], x = wh[j];
            acc += w.x * x.x + w.y * x.y + w.z * x.z + w.w * x.w;
        }
        sh_out[a] = acc;
    }
    __syncthreads();

    // ---------- Phase H: final[i] = sum_a Wo[i,a] * out[a] ----------
    if (tid < D) {
        float acc = 0.f;
        const float4* wr = (const float4*)(Wo + tid * D);
        const float4* ov = (const float4*)sh_out;
        for (int a = 0; a < D / 4; ++a) {
            float4 w = wr[a], x = ov[a];
            acc += w.x * x.x + w.y * x.y + w.z * x.z + w.w * x.w;
        }
        outp[(size_t)b * D + tid] = acc;
    }
}

extern "C" void kernel_launch(void* const* d_in, const int* in_sizes, int n_in,
                              void* d_out, int out_size, void* d_ws, size_t ws_size,
                              hipStream_t stream) {
    const float* target = (const float*)d_in[0];
    const float* hist   = (const float*)d_in[1];
    const int*   maskp  = (const int*)d_in[2];
    const float* Wq     = (const float*)d_in[3];
    const float* Wk     = (const float*)d_in[4];
    const float* Wv     = (const float*)d_in[5];
    const float* Wo     = (const float*)d_in[6];
    float* outp = (float*)d_out;
    const int B = in_sizes[0] / D;   // 4096
    mhta_kernel<<<B, 256, 0, stream>>>(target, hist, maskp, Wq, Wk, Wv, Wo, outp);
}
